// MyNet_3599182594621
// MI455X (gfx1250) — matrix operand order repaired and checked by reading, not compiled, not run on hardware
//
#include <hip/hip_runtime.h>
#include <hip/hip_bf16.h>
#include <math.h>

// ---------------------------------------------------------------------------
// MI455X (gfx1250, wave32) implementation of the Mamba-UNet forward pass.
//  - channel-last activations (B, V, C) so GEMM K-reads are contiguous
//  - all channel-mixing matmuls -> v_wmma_f32_16x16x32_f16 (f32 accumulate)
//  - weights pre-converted to f16 ONCE per launch (amortized over up to 32768
//    activation rows); B fragments load as 16 contiguous halves (32B)
//  - branchless WMMA path: K%32==0 always; OOB rows/bias index-clamped
//    (their products only reach store-guarded outputs) -> no EXEC toggling
//  - conv biases before InstanceNorm eliminated (mathematically exact)
//  - Mamba selective scan: 3-pass chunked parallel scan, pass3 fuses
//    y = h.C + x*D and the SiLU gate with 16-lane shfl reductions
// ---------------------------------------------------------------------------

#define ACT_NONE     0
#define ACT_LRELU    1
#define ACT_SOFTPLUS 2

#define PRS 576   // padded row stride (floats) for hypernet params: 16B-aligned rows

typedef __attribute__((ext_vector_type(16))) _Float16 v16h;
typedef __attribute__((ext_vector_type(8)))  float    v8f;

static __device__ __forceinline__ float lrelu_f(float v){ return v >= 0.f ? v : 0.01f * v; }
static __device__ __forceinline__ float silu_f(float v){ return v / (1.f + __expf(-v)); }

// ---------------- WMMA fragment loaders -----------------------------------
// A (16x32 f16): lane m = lane&15, kgrp=(lane>>4)*16, k(i) = kgrp + i
// B (32x16 f16): lane n = lane&15, kb16=(lane>>4)*16, k(i) = kb16 + i
__device__ __forceinline__ v16h load_a_frag(const float* row, int kgrp){
  const float4* p0 = (const float4*)(row + kgrp);
  const float4* p1 = (const float4*)(row + kgrp + 8);
  float4 a0 = p0[0], a1 = p0[1], b0 = p1[0], b1 = p1[1];
  v16h r;
  r[0]=(_Float16)a0.x; r[1]=(_Float16)a0.y; r[2]=(_Float16)a0.z; r[3]=(_Float16)a0.w;
  r[4]=(_Float16)a1.x; r[5]=(_Float16)a1.y; r[6]=(_Float16)a1.z; r[7]=(_Float16)a1.w;
  r[8]=(_Float16)b0.x; r[9]=(_Float16)b0.y; r[10]=(_Float16)b0.z; r[11]=(_Float16)b0.w;
  r[12]=(_Float16)b1.x; r[13]=(_Float16)b1.y; r[14]=(_Float16)b1.z; r[15]=(_Float16)b1.w;
  return r;
}
__device__ __forceinline__ v16h load_b_frag(const _Float16* row, int kb16){
  // 16 contiguous halves; 32B, 64B-aligned (K%32==0, base 256B-aligned)
  return *(const v16h*)(row + kb16);
}

// out[M,N] = act(A[M,K](f32) . W[N,K](f16)^T + bias[N])   (f32 accumulate)
// Host contract: K%32==0, lda%4==0, base pointers 16B-aligned.
// Each wave computes a 16x32 output tile (A fragment reused across 2 WMMAs).
__global__ void k_gemm_wmma(const float* __restrict__ A, int lda, long aB,
                            const _Float16* __restrict__ W, long wB,
                            const float* __restrict__ bias, long bB,
                            float* __restrict__ out, int ldo, long oB,
                            int M, int K, int N, int act)
{
  int lane  = threadIdx.x & 31;
  int wave  = threadIdx.x >> 5;
  int mtile = blockIdx.y * 8 + wave;
  if (mtile * 16 >= M) return;               // wave-uniform: EXEC stays full
  int b = blockIdx.z;
  A   += (size_t)b * aB;
  W   += (size_t)b * wB;
  out += (size_t)b * oB;

  int mbase = mtile * 16;
  int nbase = blockIdx.x * 32;
  int mloc  = lane & 15;
  int kgrp  = (lane >> 4) * 16;
  int kb16  = (lane >> 4) * 16;

  // clamp row indices: OOB rows feed only store-guarded outputs, so no zeroing
  int arow = mbase + mloc;       if (arow >= M) arow = M - 1;
  int wr0  = nbase + mloc;       if (wr0  >= N) wr0  = N - 1;
  int wr1  = nbase + 16 + mloc;  if (wr1  >= N) wr1  = N - 1;

  const float*    ap  = A + (size_t)arow * lda;
  const _Float16* wp0 = W + (size_t)wr0  * K;
  const _Float16* wp1 = W + (size_t)wr1  * K;

  v8f acc0 = {}, acc1 = {};
  for (int k0 = 0; k0 < K; k0 += 32){
    v16h av = load_a_frag(ap  + k0, kgrp);
    v16h b0 = load_b_frag(wp0 + k0, kb16);
    v16h b1 = load_b_frag(wp1 + k0, kb16);
    acc0 = __builtin_amdgcn_wmma_f32_16x16x32_f16(false, av, false, b0,
                                                  (short)0, acc0, false, false);
    acc1 = __builtin_amdgcn_wmma_f32_16x16x32_f16(false, av, false, b1,
                                                  (short)0, acc1, false, false);
  }

  int n0 = nbase + (lane & 15);
  int n1 = n0 + 16;
  int mo = mbase + (lane >> 4) * 8;
  float bv0 = 0.f, bv1 = 0.f;
  if (bias){                                   // uniform branch; clamped loads
    const float* bp = bias + (size_t)b * bB;
    bv0 = bp[n0 < N ? n0 : N - 1];
    bv1 = bp[n1 < N ? n1 : N - 1];
  }
#pragma unroll
  for (int r = 0; r < 8; ++r){
    int m = mo + r;
    if (m < M){
      if (n0 < N){
        float v = acc0[r] + bv0;
        if (act == ACT_LRELU) v = lrelu_f(v);
        out[(size_t)m * ldo + n0] = v;
      }
      if (n1 < N){
        float v = acc1[r] + bv1;
        if (act == ACT_LRELU) v = lrelu_f(v);
        out[(size_t)m * ldo + n1] = v;
      }
    }
  }
}

// weight pre-conversion: f32 -> f16 (amortized once per launch)
__global__ void k_f32_to_f16(const float* __restrict__ src, _Float16* __restrict__ dst,
                             long n)
{
  long i = blockIdx.x * (long)blockDim.x + threadIdx.x;
  if (i < n) dst[i] = (_Float16)src[i];
}
__global__ void k_f32_to_f16_rows(const float* __restrict__ src, long srcStride,
                                  _Float16* __restrict__ dst, long dstStride,
                                  int rowLen, int rows)
{
  long i = blockIdx.x * (long)blockDim.x + threadIdx.x;
  if (i >= (long)rowLen * rows) return;
  long r = i / rowLen;
  int  c = (int)(i % rowLen);
  dst[r * dstStride + c] = (_Float16)src[r * srcStride + c];
}

// Naive linear for tiny K or N (K=1 first conv, dt projection, ctrl head).
__global__ void k_linear_naive(const float* __restrict__ A, int lda, long aB,
                               const float* __restrict__ W, long wB,
                               const float* __restrict__ bias, long bB,
                               float* __restrict__ out, int ldo, long oB,
                               int M, int K, int N, int act)
{
  long t = blockIdx.x * (long)blockDim.x + threadIdx.x;
  if (t >= (long)M * N) return;
  int  n = (int)(t % N);
  long m = t / N;
  int  b = blockIdx.z;
  const float* ap = A + (size_t)b * aB + (size_t)m * lda;
  const float* wp = W + (size_t)b * wB + (size_t)n * K;
  float acc = bias ? bias[(size_t)b * bB + n] : 0.f;
  for (int k = 0; k < K; ++k) acc += ap[k] * wp[k];
  if      (act == ACT_LRELU)    acc = lrelu_f(acc);
  else if (act == ACT_SOFTPLUS) acc = (acc > 20.f) ? acc : log1pf(__expf(acc));
  out[(size_t)b * oB + (size_t)m * ldo + n] = acc;
}

// ---------------- normalization ----------------
__global__ void k_inorm_stats(const float* __restrict__ x, float* __restrict__ stats,
                              int V, int C)
{
  int bc = blockIdx.x;
  int b = bc / C, c = bc % C;
  int tid = threadIdx.x;
  float s = 0.f, s2 = 0.f;
  for (int v = tid; v < V; v += blockDim.x){
    float t = x[((size_t)b * V + v) * C + c];
    s += t; s2 += t * t;
  }
  __shared__ float sh[256], sh2[256];
  sh[tid] = s; sh2[tid] = s2; __syncthreads();
  for (int o = 128; o; o >>= 1){
    if (tid < o){ sh[tid] += sh[tid + o]; sh2[tid] += sh2[tid + o]; }
    __syncthreads();
  }
  if (tid == 0){
    float mean = sh[0] / V;
    float var  = sh2[0] / V - mean * mean;
    stats[2 * bc]     = mean;
    stats[2 * bc + 1] = rsqrtf(var + 1e-5f);
  }
}
__global__ void k_inorm_apply(const float* __restrict__ x, const float* __restrict__ stats,
                              const float* __restrict__ res, float* __restrict__ out,
                              long total, int V, int C)
{
  long idx = blockIdx.x * (long)blockDim.x + threadIdx.x;
  if (idx >= total) return;
  int  c  = (int)(idx % C);
  long bc = (idx / ((long)V * C)) * C + c;
  float v = (x[idx] - stats[2 * bc]) * stats[2 * bc + 1];
  v = lrelu_f(v);
  if (res) v += res[idx];
  out[idx] = v;
}
__global__ void k_layernorm(const float* __restrict__ x, const float* __restrict__ g,
                            const float* __restrict__ bta, float* __restrict__ out,
                            int C, long ntok)
{
  long tok = blockIdx.x * 8L + (threadIdx.x >> 5);
  if (tok >= ntok) return;                   // wave-uniform
  int lane = threadIdx.x & 31;
  const float* xp = x + tok * (size_t)C;
  float s = 0.f, s2 = 0.f;
  for (int c = lane; c < C; c += 32){ float t = xp[c]; s += t; s2 += t * t; }
  for (int m = 16; m; m >>= 1){ s += __shfl_xor(s, m, 32); s2 += __shfl_xor(s2, m, 32); }
  float mean = s / C;
  float rstd = rsqrtf(s2 / C - mean * mean + 1e-5f);
  float* op = out + tok * (size_t)C;
  for (int c = lane; c < C; c += 32) op[c] = (xp[c] - mean) * rstd * g[c] + bta[c];
}

// ---------------- spatial ops ----------------
__global__ void k_conv3g(const float* __restrict__ in, const float* __restrict__ w,
                         float* __restrict__ out, long total,
                         int C, int D, int H, int W, int cpg)
{
  long idx = blockIdx.x * (long)blockDim.x + threadIdx.x;
  if (idx >= total) return;
  int  c = (int)(idx % C); long t = idx / C;
  int  x = (int)(t % W); t /= W;
  int  y = (int)(t % H); t /= H;
  int  z = (int)(t % D); long b = t / D;
  int cb = (c / cpg) * cpg;
  long V = (long)D * H * W;
  const float* wbase = w + (size_t)c * cpg * 27;
  float acc = 0.f;
  for (int dz = -1; dz <= 1; ++dz){ int zz = z + dz; if (zz < 0 || zz >= D) continue;
    for (int dy = -1; dy <= 1; ++dy){ int yy = y + dy; if (yy < 0 || yy >= H) continue;
      for (int dx = -1; dx <= 1; ++dx){ int xx = x + dx; if (xx < 0 || xx >= W) continue;
        int tap = ((dz + 1) * 3 + (dy + 1)) * 3 + (dx + 1);
        const float* ip = in + ((size_t)b * V + (((long)zz * H + yy) * W + xx)) * C + cb;
        for (int ci = 0; ci < cpg; ++ci) acc += ip[ci] * wbase[(size_t)ci * 27 + tap];
      }
    }
  }
  out[idx] = acc;   // bias dropped: InstanceNorm follows (exact)
}
__global__ void k_maxpool(const float* __restrict__ in, float* __restrict__ out,
                          long total, int C, int D, int H, int W)
{
  long idx = blockIdx.x * (long)blockDim.x + threadIdx.x;
  if (idx >= total) return;
  int Do = D / 2, Ho = H / 2, Wo = W / 2;
  int c = (int)(idx % C); long t = idx / C;
  int xo = (int)(t % Wo); t /= Wo;
  int yo = (int)(t % Ho); t /= Ho;
  int zo = (int)(t % Do); long b = t / Do;
  float m = -3.402823466e38f;
  for (int dz = 0; dz < 2; ++dz)
    for (int dy = 0; dy < 2; ++dy)
      for (int dx = 0; dx < 2; ++dx){
        long v = ((long)(2 * zo + dz) * H + (2 * yo + dy)) * W + (2 * xo + dx);
        m = fmaxf(m, in[((size_t)b * D * H * W + v) * C + c]);
      }
  out[idx] = m;
}
__global__ void k_upsample(const float* __restrict__ x, const float* __restrict__ w,
                           const float* __restrict__ bias, float* __restrict__ out,
                           long total, int Cin, int Co, int Di, int Hi, int Wi,
                           int coff, int Ctot)
{
  long idx = blockIdx.x * (long)blockDim.x + threadIdx.x;
  if (idx >= total) return;
  int p = (int)(idx % Co); long t = idx / Co;
  int Do = 2 * Di, Ho = 2 * Hi, Wo = 2 * Wi;
  int xo = (int)(t % Wo); t /= Wo;
  int yo = (int)(t % Ho); t /= Ho;
  int zo = (int)(t % Do); long b = t / Do;
  int par = ((zo & 1) * 2 + (yo & 1)) * 2 + (xo & 1);
  long Vi = (long)Di * Hi * Wi;
  const float* xp = x + ((size_t)b * Vi + (((long)(zo >> 1) * Hi + (yo >> 1)) * Wi + (xo >> 1))) * Cin;
  const float* wp = w + (size_t)p * 8 + par;   // up_w (Cin, Co, 2,2,2)
  float acc = bias[p];
  for (int c = 0; c < Cin; ++c) acc += xp[c] * wp[(size_t)c * Co * 8];
  long Vo = (long)Do * Ho * Wo;
  out[((size_t)b * Vo + (((long)zo * Ho + yo) * Wo + xo)) * Ctot + coff + p] = acc;
}
__global__ void k_copy_ch(const float* __restrict__ src, float* __restrict__ dst,
                          long total, int Cs, int Ctot, int coff)
{
  long idx = blockIdx.x * (long)blockDim.x + threadIdx.x;
  if (idx >= total) return;
  int c = (int)(idx % Cs); long t = idx / Cs;
  dst[(size_t)t * Ctot + coff + c] = src[idx];
}

// ---------------- Mamba pieces ----------------
__global__ void k_dwconv_silu(const float* __restrict__ xz, const float* __restrict__ w,
                              const float* __restrict__ bias, float* __restrict__ out,
                              long total, int di, int L)
{
  long idx = blockIdx.x * (long)blockDim.x + threadIdx.x;
  if (idx >= total) return;
  int c = (int)(idx % di); long t = idx / di;
  int l = (int)(t % L); long b = t / L;
  float acc = bias[c];
#pragma unroll
  for (int j = 0; j < 4; ++j){
    int ls = l - 3 + j;
    if (ls >= 0) acc += xz[((size_t)b * L + ls) * (2 * di) + c] * w[c * 4 + j];
  }
  out[idx] = silu_f(acc);
}

// chunked selective scan: h[l] = exp(delta*A)*h[l-1] + delta*B*x
__global__ void k_scan_pass1(const float* __restrict__ delta, const float* __restrict__ xm,
                             const float* __restrict__ xdbl, int sx, int dr,
                             const float* __restrict__ A_log, int di, int L, int chunk,
                             float* __restrict__ aprod, float* __restrict__ hend)
{
  int cs = blockIdx.x * blockDim.x + threadIdx.x;
  int CS = di * 16;
  if (cs >= CS) return;
  int b = blockIdx.z, ck = blockIdx.y;
  int c = cs >> 4, s = cs & 15;
  float Acs = -__expf(A_log[c * 16 + s]);
  int l0 = ck * chunk;
  int l1 = l0 + chunk; if (l1 > L) l1 = L;
  float ap = 1.f, h = 0.f;
  for (int l = l0; l < l1; ++l){
    size_t tok = (size_t)b * L + l;
    float dlt = delta[tok * di + c];
    float dA  = __expf(dlt * Acs);
    float bx  = dlt * xdbl[tok * sx + dr + s] * xm[tok * di + c];
    h  = dA * h + bx;
    ap = ap * dA;
  }
  size_t o = ((size_t)b * gridDim.y + ck) * CS + cs;
  aprod[o] = ap; hend[o] = h;
}
__global__ void k_scan_pass2(const float* __restrict__ aprod, const float* __restrict__ hend,
                             float* __restrict__ carry, int CS, int nch)
{
  int cs = blockIdx.x * blockDim.x + threadIdx.x;
  if (cs >= CS) return;
  int b = blockIdx.z;
  float h = 0.f;
  for (int ck = 0; ck < nch; ++ck){
    size_t o = ((size_t)b * nch + ck) * CS + cs;
    carry[o] = h;
    h = aprod[o] * h + hend[o];
  }
}
__global__ void k_scan_pass3(const float* __restrict__ delta, const float* __restrict__ xm,
                             const float* __restrict__ xdbl, int sx, int dr,
                             const float* __restrict__ A_log, const float* __restrict__ Dvec,
                             const float* __restrict__ xz, const float* __restrict__ carry,
                             int di, int L, int chunk, float* __restrict__ ygate)
{
  int cs = blockIdx.x * blockDim.x + threadIdx.x;
  int CS = di * 16;
  int b = blockIdx.z, ck = blockIdx.y;
  int c = cs >> 4, s = cs & 15;
  bool active = cs < CS;             // CS is a multiple of 256; kept for safety
  float Acs = active ? -__expf(A_log[c * 16 + s]) : 0.f;
  float h   = active ? carry[((size_t)b * gridDim.y + ck) * CS + cs] : 0.f;
  int l0 = ck * chunk;
  int l1 = l0 + chunk; if (l1 > L) l1 = L;
  for (int l = l0; l < l1; ++l){
    float y = 0.f, xmv = 0.f;
    size_t tok = (size_t)b * L + l;
    if (active){
      float dlt = delta[tok * di + c];
      float dA  = __expf(dlt * Acs);
      xmv = xm[tok * di + c];
      h = dA * h + dlt * xdbl[tok * sx + dr + s] * xmv;
      y = h * xdbl[tok * sx + dr + 16 + s];        // Cm
    }
    for (int m = 8; m; m >>= 1) y += __shfl_xor(y, m, 16);   // sum over 16 states
    if (active && s == 0){
      float z = xz[tok * 2 * di + di + c];
      ygate[tok * di + c] = (y + xmv * Dvec[c]) * silu_f(z);
    }
  }
}

// ---------------- head ----------------
__global__ void k_gap_onehot(const float* __restrict__ x5, const int* __restrict__ task,
                             float* __restrict__ cond, int C, int V, int B)
{
  int i = blockIdx.x * blockDim.x + threadIdx.x;
  if (i >= B * (C + 2)) return;
  int b = i / (C + 2), c = i % (C + 2);
  if (c < C){
    float s = 0.f;
    for (int v = 0; v < V; ++v) s += x5[((size_t)b * V + v) * C + c];
    cond[i] = s / V;
  } else {
    cond[i] = (task[b] == (c - C)) ? 1.f : 0.f;
  }
}
__global__ void k_logits(const float* __restrict__ hf, const float* __restrict__ pr,
                         float* __restrict__ out, long total, int V)
{
  long idx = blockIdx.x * (long)blockDim.x + threadIdx.x;
  if (idx >= total) return;
  int  o = (int)(idx % 2); long t = idx / 2;
  long v = t % V; long b = t / V;
  const float* w2 = pr + b * PRS + 512 + o * 16;
  const float* hp = hf + ((size_t)b * V + v) * 16;
  float acc = pr[b * PRS + 560 + o];
#pragma unroll
  for (int k = 0; k < 16; ++k) acc += hp[k] * w2[k];
  out[((size_t)b * 2 + o) * V + v] = acc;   // channel-first output
}

// ============================ host side ====================================
struct Arena { char* base; size_t off; };
static float* aalloc(Arena& a, size_t nElems){
  float* r = (float*)(a.base + a.off);
  a.off += ((nElems * sizeof(float)) + 255) & ~(size_t)255;
  return r;
}
static _Float16* halloc(Arena& a, size_t nElems){
  _Float16* r = (_Float16*)(a.base + a.off);
  a.off += ((nElems * sizeof(_Float16)) + 255) & ~(size_t)255;
  return r;
}
static inline dim3 grid1(long n){ return dim3((unsigned)((n + 255) / 256)); }

struct DCP {
  const float *w1,*b1,*w2,*b2,*w3,*b3,*ln_g,*ln_b,*in_proj,*conv_w,*conv_b,
              *x_proj,*dt_w,*dt_b,*A_log,*Dvec,*out_proj,*up_w,*up_b;
  int cin, cout;
};
struct F16W { _Float16 *w1, *w3, *inp, *xp, *outp; };

static DCP load_dc(void* const* d_in, int& pi, int cin, int cout, bool up){
  DCP p{};
  const float** f = &p.w1;
  for (int i = 0; i < 17; ++i) f[i] = (const float*)d_in[pi++];
  if (up){ p.up_w = (const float*)d_in[pi++]; p.up_b = (const float*)d_in[pi++]; }
  p.cin = cin; p.cout = cout;
  return p;
}

static void cvt16(hipStream_t st, const float* s, _Float16* d, long n){
  k_f32_to_f16<<<grid1(n), 256, 0, st>>>(s, d, n);
}
static void gemm(hipStream_t st, const float* A, int lda, long aB,
                 const _Float16* W, long wB, const float* bias, long bB,
                 float* out, int ldo, long oB, int M, int K, int N, int act, int batches){
  // contract: K%32==0, lda%4==0
  dim3 g((N + 31) / 32, ((M + 15) / 16 + 7) / 8, batches);
  k_gemm_wmma<<<g, 256, 0, st>>>(A, lda, aB, W, wB, bias, bB, out, ldo, oB, M, K, N, act);
}
static void linear(hipStream_t st, const float* A, int lda, long aB,
                   const float* W, long wB, const float* bias, long bB,
                   float* out, int ldo, long oB, int M, int K, int N, int act, int batches){
  dim3 g((unsigned)(((long)M * N + 255) / 256), 1, batches);
  k_linear_naive<<<g, 256, 0, st>>>(A, lda, aB, W, wB, bias, bB, out, ldo, oB, M, K, N, act);
}
static void inorm_lrelu(hipStream_t st, const float* x, float* out, const float* res,
                        float* stats, int B, int V, int C){
  k_inorm_stats<<<dim3((unsigned)(B * C)), 256, 0, st>>>(x, stats, V, C);
  long total = (long)B * V * C;
  k_inorm_apply<<<grid1(total), 256, 0, st>>>(x, stats, res, out, total, V, C);
}

static void run_dc(hipStream_t st, const DCP& p, const F16W& fw,
                   const float* in, float* out,
                   int B, int D, int H, int W, Arena tmp /* by value: reused */)
{
  const int V = D * H * W;
  const int M = B * V;
  const int cout = p.cout, cin = p.cin;
  const int di = 2 * cout;
  const int dr = (cout + 15) / 16;
  const int sx = dr + 32;

  float* t1    = aalloc(tmp, (size_t)M * cout);
  float* t2    = aalloc(tmp, (size_t)M * cout);
  float* stats = aalloc(tmp, (size_t)B * cout * 2);
  float* xz    = aalloc(tmp, (size_t)M * 2 * di);
  float* xc    = aalloc(tmp, (size_t)M * di);
  float* xd    = aalloc(tmp, (size_t)M * sx);
  float* dl    = aalloc(tmp, (size_t)M * di);
  float* yg    = aalloc(tmp, (size_t)M * di);
  const int CH  = 64;
  const int nch = (V + CH - 1) / CH;
  const int CS  = di * 16;
  float* ap = aalloc(tmp, (size_t)B * nch * CS);
  float* he = aalloc(tmp, (size_t)B * nch * CS);
  float* ca = aalloc(tmp, (size_t)B * nch * CS);

  // conv1x1 #1 (bias dropped: InstanceNorm follows)
  if (cin >= 32) gemm(st, in, cin, 0, fw.w1, 0, nullptr, 0, t1, cout, 0, M, cin, cout, ACT_NONE, 1);
  else           linear(st, in, cin, 0, p.w1, 0, nullptr, 0, t1, cout, 0, M, cin, cout, ACT_NONE, 1);
  inorm_lrelu(st, t1, t1, nullptr, stats, B, V, cout);

  // grouped 3x3x3 conv, pad 1
  { long total = (long)M * cout;
    k_conv3g<<<grid1(total), 256, 0, st>>>(t1, p.w2, t2, total, cout, D, H, W, cout / 8); }
  inorm_lrelu(st, t2, t2, nullptr, stats, B, V, cout);

  // conv1x1 #3 (+ residual when cin==cout, i.e. enc4)
  gemm(st, t2, cout, 0, fw.w3, 0, nullptr, 0, t1, cout, 0, M, cout, cout, ACT_NONE, 1);
  inorm_lrelu(st, t1, t1, (cin == cout) ? in : nullptr, stats, B, V, cout);

  // ---- mamba layer ----
  k_layernorm<<<dim3((unsigned)(((long)M + 7) / 8)), 256, 0, st>>>(t1, p.ln_g, p.ln_b, t2, cout, M);
  gemm(st, t2, cout, 0, fw.inp, 0, nullptr, 0, xz, 2 * di, 0, M, cout, 2 * di, ACT_NONE, 1);
  { long total = (long)M * di;
    k_dwconv_silu<<<grid1(total), 256, 0, st>>>(xz, p.conv_w, p.conv_b, xc, total, di, V); }
  gemm(st, xc, di, 0, fw.xp, 0, nullptr, 0, xd, sx, 0, M, di, sx, ACT_NONE, 1);
  linear(st, xd, sx, 0, p.dt_w, 0, p.dt_b, 0, dl, di, 0, M, dr, di, ACT_SOFTPLUS, 1);

  dim3 gs((unsigned)(CS / 256), (unsigned)nch, (unsigned)B);
  k_scan_pass1<<<gs, 256, 0, st>>>(dl, xc, xd, sx, dr, p.A_log, di, V, CH, ap, he);
  k_scan_pass2<<<dim3((unsigned)(CS / 256), 1, (unsigned)B), 256, 0, st>>>(ap, he, ca, CS, nch);
  k_scan_pass3<<<gs, 256, 0, st>>>(dl, xc, xd, sx, dr, p.A_log, p.Dvec, xz, ca, di, V, CH, yg);

  gemm(st, yg, di, 0, fw.outp, 0, nullptr, 0, out, cout, 0, M, di, cout, ACT_NONE, 1);
}

extern "C" void kernel_launch(void* const* d_in, const int* in_sizes, int n_in,
                              void* d_out, int out_size, void* d_ws, size_t ws_size,
                              hipStream_t stream)
{
  (void)n_in; (void)out_size; (void)ws_size;
  const float* x    = (const float*)d_in[0];
  const int*   task = (const int*)d_in[1];
  int pi = 2;
  DCP Pconv = load_dc(d_in, pi,   1,  32, false);
  DCP Penc1 = load_dc(d_in, pi,  32,  64, false);
  DCP Penc2 = load_dc(d_in, pi,  64, 128, false);
  DCP Penc3 = load_dc(d_in, pi, 128, 256, false);
  DCP Penc4 = load_dc(d_in, pi, 256, 256, false);
  DCP Pdec1 = load_dc(d_in, pi, 512, 128, true);
  DCP Pdec2 = load_dc(d_in, pi, 256,  64, true);
  DCP Pdec3 = load_dc(d_in, pi, 128,  32, true);
  DCP Pdec4 = load_dc(d_in, pi,  64,  32, true);
  const float* ctrl_w = (const float*)d_in[pi++];
  const float* ctrl_b = (const float*)d_in[pi++];

  const int B  = in_sizes[0] / (16 * 32 * 32);
  const int V0 = 16 * 32 * 32, V1 = 8 * 16 * 16, V2 = 4 * 8 * 8, V3 = 2 * 4 * 4, V4 = 1 * 2 * 2;

  Arena ar{(char*)d_ws, 0};
  float* x1    = aalloc(ar, (size_t)B * V0 * 32);
  float* x2    = aalloc(ar, (size_t)B * V1 * 64);
  float* x3    = aalloc(ar, (size_t)B * V2 * 128);
  float* x4    = aalloc(ar, (size_t)B * V3 * 256);
  float* x5    = aalloc(ar, (size_t)B * V4 * 256);
  float* p1    = aalloc(ar, (size_t)B * V1 * 32);
  float* p2    = aalloc(ar, (size_t)B * V2 * 64);
  float* p3    = aalloc(ar, (size_t)B * V3 * 128);
  float* p4    = aalloc(ar, (size_t)B * V4 * 256);
  float* cat1  = aalloc(ar, (size_t)B * V3 * 512);
  float* cat2  = aalloc(ar, (size_t)B * V2 * 256);
  float* cat3  = aalloc(ar, (size_t)B * V1 * 128);
  float* cat4  = aalloc(ar, (size_t)B * V0 * 64);
  float* m1    = aalloc(ar, (size_t)B * V3 * 128);
  float* m2    = aalloc(ar, (size_t)B * V2 * 64);
  float* m3    = aalloc(ar, (size_t)B * V1 * 32);
  float* m4    = aalloc(ar, (size_t)B * V0 * 32);
  float* hfeat = aalloc(ar, (size_t)B * V0 * 16);
  float* cond  = aalloc(ar, (size_t)B * 258);
  float* pr    = aalloc(ar, (size_t)B * PRS);   // padded rows: 16B-aligned per batch
  _Float16* hw1 = halloc(ar, (size_t)B * 512);  // hypernet per-sample w1, f16

  // f16 weight mirrors (converted once per launch, amortized over all rows)
  DCP* dcs[9] = {&Pconv,&Penc1,&Penc2,&Penc3,&Penc4,&Pdec1,&Pdec2,&Pdec3,&Pdec4};
  F16W fw[9];
  for (int i = 0; i < 9; ++i){
    const DCP& p = *dcs[i];
    int cout = p.cout, cin = p.cin, di = 2 * cout, dr = (cout + 15) / 16, sx = dr + 32;
    F16W w{};
    if (cin >= 32){ w.w1 = halloc(ar, (size_t)cout * cin);
                    cvt16(stream, p.w1, w.w1, (long)cout * cin); }
    w.w3   = halloc(ar, (size_t)cout * cout);
    w.inp  = halloc(ar, (size_t)2 * di * cout);
    w.xp   = halloc(ar, (size_t)sx * di);
    w.outp = halloc(ar, (size_t)cout * di);
    cvt16(stream, p.w3,       w.w3,   (long)cout * cout);
    cvt16(stream, p.in_proj,  w.inp,  (long)2 * di * cout);
    cvt16(stream, p.x_proj,   w.xp,   (long)sx * di);
    cvt16(stream, p.out_proj, w.outp, (long)cout * di);
    fw[i] = w;
  }
  Arena tmp = ar;   // transient arena, reused by every double_conv

  // ---- encoder ---- (x is (B,V0,1): channel-last with C=1 == NCDHW layout)
  run_dc(stream, Pconv, fw[0], x, x1, B, 16, 32, 32, tmp);
  k_maxpool<<<grid1((long)B * V1 * 32), 256, 0, stream>>>(x1, p1, (long)B * V1 * 32, 32, 16, 32, 32);
  run_dc(stream, Penc1, fw[1], p1, x2, B, 8, 16, 16, tmp);
  k_maxpool<<<grid1((long)B * V2 * 64), 256, 0, stream>>>(x2, p2, (long)B * V2 * 64, 64, 8, 16, 16);
  run_dc(stream, Penc2, fw[2], p2, x3, B, 4, 8, 8, tmp);
  k_maxpool<<<grid1((long)B * V3 * 128), 256, 0, stream>>>(x3, p3, (long)B * V3 * 128, 128, 4, 8, 8);
  run_dc(stream, Penc3, fw[3], p3, x4, B, 2, 4, 4, tmp);
  k_maxpool<<<grid1((long)B * V4 * 256), 256, 0, stream>>>(x4, p4, (long)B * V4 * 256, 256, 2, 4, 4);
  run_dc(stream, Penc4, fw[4], p4, x5, B, 1, 2, 2, tmp);

  // ---- decoder: concat([skip, upsample(x)], channel axis) then double_conv
  k_copy_ch <<<grid1((long)B * V3 * 256), 256, 0, stream>>>(x4, cat1, (long)B * V3 * 256, 256, 512, 0);
  k_upsample<<<grid1((long)B * V3 * 256), 256, 0, stream>>>(x5, Pdec1.up_w, Pdec1.up_b, cat1,
                                                            (long)B * V3 * 256, 256, 256, 1, 2, 2, 256, 512);
  run_dc(stream, Pdec1, fw[5], cat1, m1, B, 2, 4, 4, tmp);

  k_copy_ch <<<grid1((long)B * V2 * 128), 256, 0, stream>>>(x3, cat2, (long)B * V2 * 128, 128, 256, 0);
  k_upsample<<<grid1((long)B * V2 * 128), 256, 0, stream>>>(m1, Pdec2.up_w, Pdec2.up_b, cat2,
                                                            (long)B * V2 * 128, 128, 128, 2, 4, 4, 128, 256);
  run_dc(stream, Pdec2, fw[6], cat2, m2, B, 4, 8, 8, tmp);

  k_copy_ch <<<grid1((long)B * V1 * 64), 256, 0, stream>>>(x2, cat3, (long)B * V1 * 64, 64, 128, 0);
  k_upsample<<<grid1((long)B * V1 * 64), 256, 0, stream>>>(m2, Pdec3.up_w, Pdec3.up_b, cat3,
                                                           (long)B * V1 * 64, 64, 64, 4, 8, 8, 64, 128);
  run_dc(stream, Pdec3, fw[7], cat3, m3, B, 8, 16, 16, tmp);

  k_copy_ch <<<grid1((long)B * V0 * 32), 256, 0, stream>>>(x1, cat4, (long)B * V0 * 32, 32, 64, 0);
  k_upsample<<<grid1((long)B * V0 * 32), 256, 0, stream>>>(m3, Pdec4.up_w, Pdec4.up_b, cat4,
                                                           (long)B * V0 * 32, 32, 32, 8, 16, 16, 32, 64);
  run_dc(stream, Pdec4, fw[8], cat4, m4, B, 16, 32, 32, tmp);

  // ---- hypernetwork head ----
  k_gap_onehot<<<grid1((long)B * 258), 256, 0, stream>>>(x5, task, cond, 256, V4, B);
  linear(stream, cond, 258, 0, ctrl_w, 0, ctrl_b, 0, pr, PRS, 0, B, 258, 562, ACT_NONE, 1);
  // per-sample w1 rows (16x32 per batch) -> f16, then WMMA
  k_f32_to_f16_rows<<<grid1((long)B * 512), 256, 0, stream>>>(pr, PRS, hw1, 512, 512, B);
  gemm(stream, m4, 32, (long)V0 * 32, hw1, 512, pr + 544, PRS,
       hfeat, 16, (long)V0 * 16, V0, 32, 16, ACT_LRELU, B);
  { long total = (long)B * V0 * 2;
    k_logits<<<grid1(total), 256, 0, stream>>>(hfeat, pr, (float*)d_out, total, V0); }
}
